// LiquidNeuralNetwork_17222818857214
// MI455X (gfx1250) — matrix operand order repaired and checked by reading, not compiled, not run on hardware
//
#include <hip/hip_runtime.h>

// ---------------- problem constants (match reference) ----------------
constexpr int V_ = 32000;
constexpr int E_ = 512;
constexpr int H_ = 512;
constexpr int S_ = 1024;
constexpr int B_ = 4;
constexpr int NH_ = 8;
constexpr int HD_ = H_ / NH_;   // 64
constexpr int F3_ = 3 * H_;     // 1536

// ---------------- WMMA fragment types ----------------
typedef __attribute__((ext_vector_type(16))) __bf16 v16bf;
typedef __attribute__((ext_vector_type(8)))  __bf16 v8bf;
typedef __attribute__((ext_vector_type(8)))  float  v8f;

__device__ __forceinline__ v8f wmma_bf16(v16bf a, v16bf b, v8f c) {
  // D = A(16x32 bf16) * B(32x16 bf16) + C(16x16 f32)
  return __builtin_amdgcn_wmma_f32_16x16x32_bf16(
      /*neg_a=*/false, a, /*neg_b=*/false, b,
      /*c_mod=*/(short)0, c, /*reuse_a=*/false, /*reuse_b=*/false);
}

// A fragment (16x32, 16-bit): lane m = lane&15, half = lane>>4.
// elements 0..7  <- A[m, kbase + half*16 + j]
// elements 8..15 <- A[m, kbase + half*16 + 8 + j]
__device__ __forceinline__ v16bf load_a_frag(const __bf16* rowptr, int kbase, int half) {
  const v8bf lo = *(const v8bf*)(rowptr + kbase + half * 16);
  const v8bf hi = *(const v8bf*)(rowptr + kbase + half * 16 + 8);
  v16bf a;
#pragma unroll
  for (int i = 0; i < 8; ++i) { a[i] = lo[i]; a[i + 8] = hi[i]; }
  return a;
}

// B fragment (32x16, 16-bit): lane col n = lane&15, K = half*16 + j
// -> one contiguous 32B load from the weight row.
__device__ __forceinline__ v16bf load_b_frag(const __bf16* __restrict__ W, int ldw,
                                             int col, int k, int half) {
  return *(const v16bf*)(W + (size_t)col * ldw + k + half * 16);
}

// ---------------- elementwise / gather kernels ----------------
__global__ void embed_pe_kernel(const int* __restrict__ tokens,
                                const float* __restrict__ emb,
                                float* __restrict__ x) {
  int i = blockIdx.x * blockDim.x + threadIdx.x;
  const int total = B_ * S_ * E_;
  if (i >= total) return;
  int e  = i % E_;
  int bs = i / E_;
  int s  = bs % S_;
  int tok = tokens[bs];
  int i2 = e >> 1;
  float div = __expf((float)(2 * i2) * (-9.210340371976184f / (float)E_)); // ln(10000)
  float arg = (float)s * div;
  float pe = (e & 1) ? __cosf(arg) : __sinf(arg);
  x[i] = emb[(size_t)tok * E_ + e] + pe;
}

__global__ void f32_to_bf16_kernel(const float* __restrict__ in,
                                   __bf16* __restrict__ out, int n) {
  int i = blockIdx.x * blockDim.x + threadIdx.x;
  if (i < n) out[i] = (__bf16)in[i];
}

// ---------------- liquid scan (serial over S, parallel over B*H) ----------------
__global__ void liquid_scan_kernel(const float* __restrict__ I,
                                   const float* __restrict__ tau,
                                   const float* __restrict__ log_dt,
                                   float* __restrict__ sraw) {
  int t = blockIdx.x * blockDim.x + threadIdx.x;
  if (t >= B_ * H_) return;
  int b = t / H_, h = t % H_;
  float dt = __expf(log_dt[0]);
  float r = dt / tau[h];
  float u = 0.f, s = 0.f;
  const float* Ip = I + (size_t)b * S_ * H_ + h;
  float* op = sraw + (size_t)b * S_ * H_ + h;
  for (int step = 0; step < S_; ++step) {
    float It = Ip[(size_t)step * H_];
    float un = u + r * (It - u);
    float sn = s + r * (u - s);   // uses OLD u (matches scan carry semantics)
    op[(size_t)step * H_] = sn;
    u = un; s = sn;
  }
}

// ---------------- tanh + LayerNorm over H per (b,s) row ----------------
__global__ void __launch_bounds__(256)
tanh_layernorm_kernel(const float* __restrict__ in,
                      const float* __restrict__ g,
                      const float* __restrict__ beta,
                      float* __restrict__ out) {
  __shared__ float ssum[8], ssq[8];
  int row = blockIdx.x;                       // 0 .. B*S-1
  const float* ip = in + (size_t)row * H_;
  float v0 = tanhf(ip[threadIdx.x]);
  float v1 = tanhf(ip[threadIdx.x + 256]);
  float sum = v0 + v1;
  float sq  = v0 * v0 + v1 * v1;
#pragma unroll
  for (int m = 1; m < 32; m <<= 1) {
    sum += __shfl_xor(sum, m, 32);
    sq  += __shfl_xor(sq,  m, 32);
  }
  int wid = threadIdx.x >> 5;
  if ((threadIdx.x & 31) == 0) { ssum[wid] = sum; ssq[wid] = sq; }
  __syncthreads();
  if (threadIdx.x == 0) {
    float a = 0.f, q = 0.f;
#pragma unroll
    for (int i = 0; i < 8; ++i) { a += ssum[i]; q += ssq[i]; }
    ssum[0] = a; ssq[0] = q;
  }
  __syncthreads();
  float mean = ssum[0] * (1.0f / H_);
  float var  = ssq[0] * (1.0f / H_) - mean * mean;
  float inv  = rsqrtf(var + 1e-5f);
  float* op = out + (size_t)row * H_;
  op[threadIdx.x]       = g[threadIdx.x]       * (v0 - mean) * inv + beta[threadIdx.x];
  op[threadIdx.x + 256] = g[threadIdx.x + 256] * (v1 - mean) * inv + beta[threadIdx.x + 256];
}

// ---------------- generic WMMA GEMM: C[M,N] = A[M,K] * W[N,K]^T + bias ----------------
// One wave computes a 32x64 output tile (2 A-frags x 4 B-frags = 8 WMMAs/k-step).
// Software-pipelined: next k-chunk fragments are loaded before the current
// chunk's WMMAs execute, so loads overlap matrix-pipe work.
// Grid: (N/64, M/32).
__global__ void __launch_bounds__(32)
wmma_gemm_bias(const __bf16* __restrict__ A, int lda,
               const __bf16* __restrict__ W, int ldw,
               const float* __restrict__ bias,
               float* __restrict__ C, int ldc, int K) {
  const int lane = threadIdx.x;
  const int n16  = lane & 15;
  const int half = lane >> 4;
  const int row0 = blockIdx.y * 32;
  const int col0 = blockIdx.x * 64;

  const __bf16* arow0 = A + (size_t)(row0 + n16) * lda;
  const __bf16* arow1 = arow0 + (size_t)16 * lda;

  v8f acc[2][4] = {};

  // prologue fragments (k = 0)
  v16bf a0 = load_a_frag(arow0, 0, half);
  v16bf a1 = load_a_frag(arow1, 0, half);
  v16bf b0 = load_b_frag(W, ldw, col0 + 0 * 16 + n16, 0, half);
  v16bf b1 = load_b_frag(W, ldw, col0 + 1 * 16 + n16, 0, half);
  v16bf b2 = load_b_frag(W, ldw, col0 + 2 * 16 + n16, 0, half);
  v16bf b3 = load_b_frag(W, ldw, col0 + 3 * 16 + n16, 0, half);

  for (int k = 32; k < K; k += 32) {
    // issue next-chunk loads first: they fly while the 8 WMMAs below execute
    v16bf na0 = load_a_frag(arow0, k, half);
    v16bf na1 = load_a_frag(arow1, k, half);
    v16bf nb0 = load_b_frag(W, ldw, col0 + 0 * 16 + n16, k, half);
    v16bf nb1 = load_b_frag(W, ldw, col0 + 1 * 16 + n16, k, half);
    v16bf nb2 = load_b_frag(W, ldw, col0 + 2 * 16 + n16, k, half);
    v16bf nb3 = load_b_frag(W, ldw, col0 + 3 * 16 + n16, k, half);
    __builtin_prefetch(arow0 + k + 32, 0, 3);

    acc[0][0] = wmma_bf16(a0, b0, acc[0][0]);
    acc[1][0] = wmma_bf16(a1, b0, acc[1][0]);
    acc[0][1] = wmma_bf16(a0, b1, acc[0][1]);
    acc[1][1] = wmma_bf16(a1, b1, acc[1][1]);
    acc[0][2] = wmma_bf16(a0, b2, acc[0][2]);
    acc[1][2] = wmma_bf16(a1, b2, acc[1][2]);
    acc[0][3] = wmma_bf16(a0, b3, acc[0][3]);
    acc[1][3] = wmma_bf16(a1, b3, acc[1][3]);

    a0 = na0; a1 = na1;
    b0 = nb0; b1 = nb1; b2 = nb2; b3 = nb3;
  }
  // epilogue: last chunk
  acc[0][0] = wmma_bf16(a0, b0, acc[0][0]);
  acc[1][0] = wmma_bf16(a1, b0, acc[1][0]);
  acc[0][1] = wmma_bf16(a0, b1, acc[0][1]);
  acc[1][1] = wmma_bf16(a1, b1, acc[1][1]);
  acc[0][2] = wmma_bf16(a0, b2, acc[0][2]);
  acc[1][2] = wmma_bf16(a1, b2, acc[1][2]);
  acc[0][3] = wmma_bf16(a0, b3, acc[0][3]);
  acc[1][3] = wmma_bf16(a1, b3, acc[1][3]);

#pragma unroll
  for (int mt = 0; mt < 2; ++mt) {
#pragma unroll
    for (int t = 0; t < 4; ++t) {
#pragma unroll
      for (int r = 0; r < 8; ++r) {
        int row = row0 + mt * 16 + half * 8 + r;
        int col = col0 + t * 16 + n16;
        float bv = bias ? bias[col] : 0.0f;
        C[(size_t)row * ldc + col] = acc[mt][t][r] + bv;
      }
    }
  }
}

// ---------------- V transpose for attention: vt[b,h,d,s] = qkv[b,s, 2H + h*64 + d] ----------------
__global__ void transpose_v_kernel(const __bf16* __restrict__ qkv,
                                   __bf16* __restrict__ vt) {
  int i = blockIdx.x * blockDim.x + threadIdx.x;
  const int total = B_ * NH_ * HD_ * S_;
  if (i >= total) return;
  int s = i % S_;
  int d = (i / S_) % HD_;
  int h = (i / (S_ * HD_)) % NH_;
  int b = i / (S_ * HD_ * NH_);
  vt[i] = qkv[(size_t)(b * S_ + s) * F3_ + 2 * H_ + h * HD_ + d];
}

// ---------------- flash attention: one wave per (b, head, 16-query tile) ----------------
__global__ void __launch_bounds__(32)
flash_attn_kernel(const __bf16* __restrict__ qkv,
                  const __bf16* __restrict__ vt,
                  float* __restrict__ out) {
  __shared__ __bf16 ptile[16][32];     // P tile staging (C-layout -> A-layout transpose)
  const int lane = threadIdx.x;
  const int n16  = lane & 15;
  const int half = lane >> 4;
  const int nqt = S_ / 16;
  const int qt = blockIdx.x % nqt;
  const int h  = (blockIdx.x / nqt) % NH_;
  const int b  = blockIdx.x / (nqt * NH_);
  const int q0 = qt * 16;

  const __bf16* Qrow  = qkv + (size_t)(b * S_ + q0 + n16) * F3_ + h * HD_;
  const __bf16* Kbase = qkv + (size_t)b * S_ * F3_ + H_ + h * HD_;
  const __bf16* Vt    = vt  + (size_t)((b * NH_ + h) * HD_) * S_;

  // Q fragments (d = 0..31 and 32..63); reused over the whole K loop
  v16bf aq0 = load_a_frag(Qrow, 0,  half);
  v16bf aq1 = load_a_frag(Qrow, 32, half);

  v8f o[4] = {};
  float mrow[8], lrow[8];
#pragma unroll
  for (int r = 0; r < 8; ++r) { mrow[r] = -1e30f; lrow[r] = 0.f; }
  const float scale = 0.125f;  // 1/sqrt(64)

  for (int k0 = 0; k0 < S_; k0 += 32) {
    // ---- scores: 16x32 = two 16x16 WMMA tiles, each K-split over d halves ----
    v8f s0 = {}, s1 = {};
    {
      const __bf16* kr0 = Kbase + (size_t)(k0 + n16) * F3_;
      const __bf16* kr1 = Kbase + (size_t)(k0 + 16 + n16) * F3_;
      v16bf b00 = *(const v16bf*)(kr0 + half * 16);       // d 0..31
      v16bf b01 = *(const v16bf*)(kr0 + 32 + half * 16);  // d 32..63
      v16bf b10 = *(const v16bf*)(kr1 + half * 16);
      v16bf b11 = *(const v16bf*)(kr1 + 32 + half * 16);
      s0 = wmma_bf16(aq0, b00, s0);
      s0 = wmma_bf16(aq1, b01, s0);
      s1 = wmma_bf16(aq0, b10, s1);
      s1 = wmma_bf16(aq1, b11, s1);
    }
    __syncthreads();   // guard ptile reuse from previous iteration
    // ---- online softmax per query row; rows of one half live in its 16 lanes ----
#pragma unroll
    for (int r = 0; r < 8; ++r) {
      float v0 = s0[r] * scale;
      float v1 = s1[r] * scale;
      float mx = fmaxf(v0, v1);
#pragma unroll
      for (int msk = 1; msk <= 8; msk <<= 1) mx = fmaxf(mx, __shfl_xor(mx, msk, 32));
      float mnew  = fmaxf(mrow[r], mx);
      float alpha = __expf(mrow[r] - mnew);
      float p0 = __expf(v0 - mnew);
      float p1 = __expf(v1 - mnew);
      float rs = p0 + p1;
#pragma unroll
      for (int msk = 1; msk <= 8; msk <<= 1) rs += __shfl_xor(rs, msk, 32);
      lrow[r] = lrow[r] * alpha + rs;
      mrow[r] = mnew;
#pragma unroll
      for (int t = 0; t < 4; ++t) o[t][r] *= alpha;
      ptile[half * 8 + r][n16]      = (__bf16)p0;
      ptile[half * 8 + r][16 + n16] = (__bf16)p1;
    }
    __syncthreads();
    // ---- P(16x32) * V(32x64): 4 WMMAs into o accumulators ----
    v16bf pa = load_a_frag(&ptile[n16][0], 0, half);
#pragma unroll
    for (int t = 0; t < 4; ++t) {
      const __bf16* vr = Vt + (size_t)(t * 16 + n16) * S_ + k0 + half * 16;
      v16bf bv = *(const v16bf*)vr;
      o[t] = wmma_bf16(pa, bv, o[t]);
    }
  }
  // ---- normalize and write head-interleaved output [B,S,H] ----
#pragma unroll
  for (int t = 0; t < 4; ++t) {
#pragma unroll
    for (int r = 0; r < 8; ++r) {
      int row = q0 + half * 8 + r;
      out[(size_t)(b * S_ + row) * H_ + h * HD_ + t * 16 + n16] = o[t][r] / lrow[r];
    }
  }
}

// ---------------- host-side orchestration ----------------
extern "C" void kernel_launch(void* const* d_in, const int* in_sizes, int n_in,
                              void* d_out, int out_size, void* d_ws, size_t ws_size,
                              hipStream_t stream) {
  (void)in_sizes; (void)n_in; (void)out_size; (void)ws_size;
  const int*   tokens = (const int*)  d_in[0];
  const float* emb    = (const float*)d_in[1];
  const float* W0     = (const float*)d_in[2];
  const float* b0     = (const float*)d_in[3];
  const float* tau0   = (const float*)d_in[4];
  const float* g0     = (const float*)d_in[5];
  const float* be0    = (const float*)d_in[6];
  const float* W1     = (const float*)d_in[7];
  const float* b1     = (const float*)d_in[8];
  const float* tau1   = (const float*)d_in[9];
  const float* g1     = (const float*)d_in[10];
  const float* be1    = (const float*)d_in[11];
  const float* w_in   = (const float*)d_in[12];
  const float* b_in   = (const float*)d_in[13];
  const float* w_o    = (const float*)d_in[14];
  const float* b_o    = (const float*)d_in[15];
  const float* Wout   = (const float*)d_in[16];
  const float* bout   = (const float*)d_in[17];
  const float* log_dt = (const float*)d_in[18];
  float* out = (float*)d_out;

  char* ws = (char*)d_ws;
  size_t off = 0;
  auto alloc = [&](size_t bytes) -> void* {
    off = (off + 255) & ~(size_t)255;
    void* p = ws + off;
    off += bytes;
    return p;
  };
  float*  buf0 = (float*) alloc((size_t)B_ * S_ * E_  * 4);  // x / scan raw
  float*  buf1 = (float*) alloc((size_t)B_ * S_ * H_  * 4);  // I / attn out
  float*  buf2 = (float*) alloc((size_t)B_ * S_ * H_  * 4);  // LN out / proj out
  float*  bufq = (float*) alloc((size_t)B_ * S_ * F3_ * 4);  // qkv f32
  __bf16* bb0  = (__bf16*)alloc((size_t)B_ * S_ * F3_ * 2);  // generic bf16 activations
  __bf16* bbq  = (__bf16*)alloc((size_t)B_ * S_ * F3_ * 2);  // qkv bf16
  __bf16* vtb  = (__bf16*)alloc((size_t)B_ * NH_ * HD_ * S_ * 2);
  __bf16* W0b  = (__bf16*)alloc((size_t)H_  * E_ * 2);
  __bf16* W1b  = (__bf16*)alloc((size_t)H_  * H_ * 2);
  __bf16* Winb = (__bf16*)alloc((size_t)F3_ * H_ * 2);
  __bf16* Wob  = (__bf16*)alloc((size_t)H_  * H_ * 2);
  __bf16* Woutb= (__bf16*)alloc((size_t)V_  * H_ * 2);

  auto cdiv = [](int a, int b) { return (a + b - 1) / b; };
  const int M = B_ * S_;   // 4096 rows
  const dim3 wblk(32, 1, 1);

  // --- weight fp32 -> bf16 conversions ---
  f32_to_bf16_kernel<<<cdiv(H_ * E_, 256), 256, 0, stream>>>(W0,   W0b,   H_ * E_);
  f32_to_bf16_kernel<<<cdiv(H_ * H_, 256), 256, 0, stream>>>(W1,   W1b,   H_ * H_);
  f32_to_bf16_kernel<<<cdiv(F3_ * H_, 256), 256, 0, stream>>>(w_in, Winb, F3_ * H_);
  f32_to_bf16_kernel<<<cdiv(H_ * H_, 256), 256, 0, stream>>>(w_o,  Wob,   H_ * H_);
  f32_to_bf16_kernel<<<cdiv(V_ * H_, 256), 256, 0, stream>>>(Wout, Woutb, V_ * H_);

  // --- embedding + positional encoding ---
  embed_pe_kernel<<<cdiv(B_ * S_ * E_, 256), 256, 0, stream>>>(tokens, emb, buf0);
  f32_to_bf16_kernel<<<cdiv(B_ * S_ * E_, 256), 256, 0, stream>>>(buf0, bb0, B_ * S_ * E_);

  // --- liquid layer 0 ---
  wmma_gemm_bias<<<dim3(H_ / 64, M / 32), wblk, 0, stream>>>(bb0, E_, W0b, E_, b0, buf1, H_, E_);
  liquid_scan_kernel<<<cdiv(B_ * H_, 256), 256, 0, stream>>>(buf1, tau0, log_dt, buf0);
  tanh_layernorm_kernel<<<M, 256, 0, stream>>>(buf0, g0, be0, buf2);
  f32_to_bf16_kernel<<<cdiv(M * H_, 256), 256, 0, stream>>>(buf2, bb0, M * H_);

  // --- liquid layer 1 ---
  wmma_gemm_bias<<<dim3(H_ / 64, M / 32), wblk, 0, stream>>>(bb0, H_, W1b, H_, b1, buf1, H_, H_);
  liquid_scan_kernel<<<cdiv(B_ * H_, 256), 256, 0, stream>>>(buf1, tau1, log_dt, buf0);
  tanh_layernorm_kernel<<<M, 256, 0, stream>>>(buf0, g1, be1, buf2);
  f32_to_bf16_kernel<<<cdiv(M * H_, 256), 256, 0, stream>>>(buf2, bb0, M * H_);

  // --- MHA: QKV projection ---
  wmma_gemm_bias<<<dim3(F3_ / 64, M / 32), wblk, 0, stream>>>(bb0, H_, Winb, H_, b_in, bufq, F3_, H_);
  f32_to_bf16_kernel<<<cdiv(M * F3_, 256), 256, 0, stream>>>(bufq, bbq, M * F3_);
  transpose_v_kernel<<<cdiv(B_ * NH_ * HD_ * S_, 256), 256, 0, stream>>>(bbq, vtb);

  // --- attention (flash-style, online softmax) ---
  flash_attn_kernel<<<B_ * NH_ * (S_ / 16), wblk, 0, stream>>>(bbq, vtb, buf1);
  f32_to_bf16_kernel<<<cdiv(M * H_, 256), 256, 0, stream>>>(buf1, bb0, M * H_);

  // --- output projection of MHA ---
  wmma_gemm_bias<<<dim3(H_ / 64, M / 32), wblk, 0, stream>>>(bb0, H_, Wob, H_, b_o, buf2, H_, H_);
  f32_to_bf16_kernel<<<cdiv(M * H_, 256), 256, 0, stream>>>(buf2, bb0, M * H_);

  // --- vocab projection (dominant GEMM): [4096,512] x [512->32000] ---
  wmma_gemm_bias<<<dim3(V_ / 64, M / 32), wblk, 0, stream>>>(bb0, H_, Woutb, H_, bout, out, V_, H_);
}
